// newmodel_42099269436064
// MI455X (gfx1250) — matrix operand order repaired and checked by reading, not compiled, not run on hardware
//
#include <hip/hip_runtime.h>

// ============================================================================
// MI455X (gfx1250) implementation, round 3.
//  - GEMM/conv core: v_wmma_f32_16x16x32_bf16, 64x64 block tile, 4 waves,
//    32x32 per wave => 4 WMMA per K-step; fragment-major LDS (2x ds_load_b128
//    per fragment); float4 staging on interior tiles; global_prefetch of the
//    next K-tile; cold edge paths rolled.
//  - Conv: tap-major im2col (weights pre-transposed AND pre-converted to bf16)
//    => B tiles staged with global_load_async_to_lds_b128 + s_wait_asynccnt.
//    Cin%32==0 convs compile only the straight-line tap paths (template).
//  - Elementwise / LN / recurrent scan stay fp32 VALU; mamba runs in 4
//    L2-resident chunks of 32 sequences.
// ============================================================================

typedef __attribute__((ext_vector_type(16))) __bf16 v16bf;
typedef __attribute__((ext_vector_type(8)))  __bf16 v8bf;
typedef __attribute__((ext_vector_type(8)))  float  v8f;
typedef __attribute__((ext_vector_type(4), aligned(4))) float f4u;  // 4B-aligned

__device__ __forceinline__ __bf16 f2bf(float f) {
  unsigned u = __builtin_bit_cast(unsigned, f);
  unsigned r = (u + 0x7FFFu + ((u >> 16) & 1u)) >> 16;
  unsigned short s = (unsigned short)r;
  return __builtin_bit_cast(__bf16, s);
}
__device__ __forceinline__ __bf16 bf_zero() {
  return __builtin_bit_cast(__bf16, (unsigned short)0);
}

// A fragment for (m, hi) is K = hi*16 + e (contiguous, matches Bf)
__device__ __forceinline__ void store_a_frag(__bf16 Af[64][40], int r, int c, float v) {
  Af[r][c] = f2bf(v);
}

// B fragment for (n, hi) is K = hi*16 + e, so Bf[n][k] is frag-major.
__device__ __forceinline__ v16bf load_frag16(const __bf16* p) {
  v8bf lo = *(const v8bf*)p;        // ds_load_b128
  v8bf hi = *(const v8bf*)(p + 8);  // ds_load_b128
  v16bf r;
#pragma unroll
  for (int i = 0; i < 8; ++i) { r[i] = lo[i]; r[8 + i] = hi[i]; }
  return r;
}

__device__ __forceinline__ float apply_act(float v, int act) {
  if (act == 1)      v = fmaxf(v, 0.0f);
  else if (act == 2) v = 0.5f * v * (1.0f + erff(v * 0.70710678118f));
  else if (act == 3) v = (v > 20.0f) ? v : log1pf(__expf(v));
  return v;
}

// ---------------------------------------------------------------------------
// Generic z-batched GEMM:  C[z] = act(op(A[z]) * op(B[z]) + bias) (+resid, +=)
// ---------------------------------------------------------------------------
struct GemmArgs {
  const float* A; const float* B; float* C;
  const float* bias; const float* resid;
  int M, N, K;
  int lda, ldb, ldc;
  long long sA, sB, sC, sR;
  int ct, biasPerM, act, accum;
};

template<int AT, int BT>
__global__ __launch_bounds__(128)
void gemm_wmma_kernel(GemmArgs g) {
  __shared__ __bf16 Af[64][40];
  __shared__ __bf16 Bf[64][40];
  const int tid  = threadIdx.x;
  const int wave = tid >> 5, lane = tid & 31;
  const int mr   = lane & 15, hi = lane >> 4;
  const int wy   = wave >> 1, wx = wave & 1;
  const int m0   = blockIdx.x * 64, n0 = blockIdx.y * 64;
  const int z    = blockIdx.z;
  const float* A = g.A + (long long)z * g.sA;
  const float* B = g.B + (long long)z * g.sB;
  float*       C = g.C + (long long)z * g.sC;
  const float* R = g.resid ? (g.resid + (long long)z * g.sR) : nullptr;
  const bool mFull = (m0 + 64 <= g.M);
  const bool nFull = (n0 + 64 <= g.N);

  v8f acc[2][2];
#pragma unroll
  for (int i = 0; i < 2; ++i)
#pragma unroll
    for (int j = 0; j < 2; ++j)
#pragma unroll
      for (int e = 0; e < 8; ++e) acc[i][j][e] = 0.0f;

  for (int k0 = 0; k0 < g.K; k0 += 32) {
    const bool kFull = (k0 + 32 <= g.K);
    // ---- stage A (64 x 32) ----
    if (mFull && kFull) {
#pragma unroll
      for (int q = 0; q < 4; ++q) {
        int chunk = tid + q * 128;
        if (AT) {
          int c = chunk >> 4, r = (chunk & 15) * 4;
          float4 v = *(const float4*)(A + (long long)(k0 + c) * g.lda + m0 + r);
          store_a_frag(Af, r + 0, c, v.x);
          store_a_frag(Af, r + 1, c, v.y);
          store_a_frag(Af, r + 2, c, v.z);
          store_a_frag(Af, r + 3, c, v.w);
        } else {
          int r = chunk >> 3, c = (chunk & 7) * 4;
          float4 v = *(const float4*)(A + (long long)(m0 + r) * g.lda + k0 + c);
          store_a_frag(Af, r, c + 0, v.x);
          store_a_frag(Af, r, c + 1, v.y);
          store_a_frag(Af, r, c + 2, v.z);
          store_a_frag(Af, r, c + 3, v.w);
        }
      }
    } else {
#pragma unroll 1
      for (int q = 0; q < 16; ++q) {
        int idx = tid + q * 128;
        int r, c;
        if (AT) { r = idx & 63; c = idx >> 6; }
        else    { c = idx & 31; r = idx >> 5; }
        int m = m0 + r, k = k0 + c;
        float v = 0.0f;
        if (m < g.M && k < g.K)
          v = AT ? A[(long long)k * g.lda + m] : A[(long long)m * g.lda + k];
        store_a_frag(Af, r, c, v);
      }
    }
    // ---- stage B (32 x 64) ----
    if (nFull && kFull) {
#pragma unroll
      for (int q = 0; q < 4; ++q) {
        int chunk = tid + q * 128;
        if (BT) {
          int n = chunk >> 3, c = (chunk & 7) * 4;
          float4 v = *(const float4*)(B + (long long)(n0 + n) * g.ldb + k0 + c);
          Bf[n][c + 0] = f2bf(v.x);
          Bf[n][c + 1] = f2bf(v.y);
          Bf[n][c + 2] = f2bf(v.z);
          Bf[n][c + 3] = f2bf(v.w);
        } else {
          int k = chunk >> 4, n = (chunk & 15) * 4;
          float4 v = *(const float4*)(B + (long long)(k0 + k) * g.ldb + n0 + n);
          Bf[n + 0][k] = f2bf(v.x);
          Bf[n + 1][k] = f2bf(v.y);
          Bf[n + 2][k] = f2bf(v.z);
          Bf[n + 3][k] = f2bf(v.w);
        }
      }
    } else {
#pragma unroll 1
      for (int q = 0; q < 16; ++q) {
        int idx = tid + q * 128;
        int n, k;
        if (BT) { k = idx & 31; n = idx >> 5; }
        else    { n = idx & 63; k = idx >> 6; }
        int kk = k0 + k, nn = n0 + n;
        float v = 0.0f;
        if (kk < g.K && nn < g.N)
          v = BT ? B[(long long)nn * g.ldb + kk] : B[(long long)kk * g.ldb + nn];
        Bf[n][k] = f2bf(v);
      }
    }
    // prefetch next K-tile (speculative; gfx1250 global_prefetch_b8)
    if (kFull && (k0 + 32 < g.K)) {
      if (tid < 64) {
        const float* pa = AT ? A + (long long)(k0 + 32 + (tid & 31)) * g.lda + m0
                             : A + (long long)(m0 + tid) * g.lda + (k0 + 32);
        __builtin_prefetch(pa, 0, 1);
      } else {
        const float* pb = BT ? B + (long long)(n0 + (tid & 63)) * g.ldb + (k0 + 32)
                             : B + (long long)(k0 + 32 + (tid & 31)) * g.ldb + n0;
        __builtin_prefetch(pb, 0, 1);
      }
    }
    __syncthreads();
    v16bf a0 = load_frag16(&Af[wy * 32      + mr][hi * 16]);
    v16bf a1 = load_frag16(&Af[wy * 32 + 16 + mr][hi * 16]);
    v16bf b0 = load_frag16(&Bf[wx * 32      + mr][hi * 16]);
    v16bf b1 = load_frag16(&Bf[wx * 32 + 16 + mr][hi * 16]);
    acc[0][0] = __builtin_amdgcn_wmma_f32_16x16x32_bf16(false, a0, false, b0, (short)0, acc[0][0], false, false);
    acc[0][1] = __builtin_amdgcn_wmma_f32_16x16x32_bf16(false, a0, false, b1, (short)0, acc[0][1], false, false);
    acc[1][0] = __builtin_amdgcn_wmma_f32_16x16x32_bf16(false, a1, false, b0, (short)0, acc[1][0], false, false);
    acc[1][1] = __builtin_amdgcn_wmma_f32_16x16x32_bf16(false, a1, false, b1, (short)0, acc[1][1], false, false);
    __syncthreads();
  }
  // ---- epilogue ----
  if (g.ct && mFull && nFull) {
#pragma unroll
    for (int ty = 0; ty < 2; ++ty)
#pragma unroll
      for (int tx = 0; tx < 2; ++tx) {
        int n  = n0 + wx * 32 + tx * 16 + mr;
        int mb = m0 + wy * 32 + ty * 16 + hi * 8;
        long long cb = (long long)n * g.ldc + mb;
        float vv[8];
        float bn = (g.bias && !g.biasPerM) ? g.bias[n] : 0.0f;
#pragma unroll
        for (int r = 0; r < 8; ++r) {
          float v = acc[ty][tx][r] + bn;
          if (g.bias && g.biasPerM) v += g.bias[mb + r];
          vv[r] = apply_act(v, g.act);
        }
        if (R) {
          float4 r0 = *(const float4*)(R + cb);
          float4 r1 = *(const float4*)(R + cb + 4);
          vv[0] += r0.x; vv[1] += r0.y; vv[2] += r0.z; vv[3] += r0.w;
          vv[4] += r1.x; vv[5] += r1.y; vv[6] += r1.z; vv[7] += r1.w;
        }
        if (g.accum) {
          float4 c0 = *(const float4*)(C + cb);
          float4 c1 = *(const float4*)(C + cb + 4);
          vv[0] += c0.x; vv[1] += c0.y; vv[2] += c0.z; vv[3] += c0.w;
          vv[4] += c1.x; vv[5] += c1.y; vv[6] += c1.z; vv[7] += c1.w;
        }
        float4 s0 = {vv[0], vv[1], vv[2], vv[3]};
        float4 s1 = {vv[4], vv[5], vv[6], vv[7]};
        *(float4*)(C + cb)     = s0;
        *(float4*)(C + cb + 4) = s1;
      }
  } else {
#pragma unroll
    for (int ty = 0; ty < 2; ++ty)
#pragma unroll
      for (int tx = 0; tx < 2; ++tx)
#pragma unroll
        for (int r = 0; r < 8; ++r) {
          int m = m0 + wy * 32 + ty * 16 + r + hi * 8;
          int n = n0 + wx * 32 + tx * 16 + mr;
          if (m < g.M && n < g.N) {
            float v = acc[ty][tx][r];
            if (g.bias) v += g.biasPerM ? g.bias[m] : g.bias[n];
            v = apply_act(v, g.act);
            long long ci = g.ct ? ((long long)n * g.ldc + m)
                                : ((long long)m * g.ldc + n);
            if (R) v += R[ci];
            if (g.accum) C[ci] += v; else C[ci] = v;
          }
        }
  }
}

// ---------------------------------------------------------------------------
// Tap-major implicit-im2col 3x3 conv (pad=1), NCHW. K = 9*Cin, k = tap*Cin+cin.
// Weights pre-transposed to bf16 Wt[n][tap][cin].
// ---------------------------------------------------------------------------
__global__ void conv_wt_kernel(const float* W, __bf16* Wt, int Cin, int cinShift,
                               int total) {
  int i = blockIdx.x * 256 + threadIdx.x;
  if (i >= total) return;
  int K = Cin * 9;
  int n = i / K, k = i - n * K;
  int t = k >> cinShift, cin = k - (t << cinShift);
  Wt[i] = f2bf(W[((long long)n * Cin + cin) * 9 + t]);
}

struct ConvArgs {
  const float* X; const __bf16* Wt; const float* bias; float* Y;
  int Cin, Cout, H, Wd, cinShift;
  long long sX, sY;
  int relu;
};

// GENERIC=0: Cin % 32 == 0 (single tap per K-tile, straight-line staging,
//            async-LDS B copies). GENERIC=1: per-element path (Cin=4 head).
template<int GENERIC>
__global__ __launch_bounds__(128)
void conv3x3_wmma_kernel(ConvArgs g) {
  __shared__ __bf16 Af[64][40];
  __shared__ __bf16 Bf[64][40];
  const int tid  = threadIdx.x;
  const int wave = tid >> 5, lane = tid & 31;
  const int mr   = lane & 15, hi = lane >> 4;
  const int wy   = wave >> 1, wx = wave & 1;
  const int M = g.H * g.Wd;
  const int K = g.Cin * 9;
  const int m0 = blockIdx.x * 64, n0 = blockIdx.y * 64;
  const float* X = g.X + (long long)blockIdx.z * g.sX;
  float*       Y = g.Y + (long long)blockIdx.z * g.sY;
  const bool nFull = (n0 + 64 <= g.Cout);
  const int y0 = m0 / g.Wd;            // one scalar division per block
  const int x0 = m0 - y0 * g.Wd;

  v8f acc[2][2];
#pragma unroll
  for (int i = 0; i < 2; ++i)
#pragma unroll
    for (int j = 0; j < 2; ++j)
#pragma unroll
      for (int e = 0; e < 8; ++e) acc[i][j][e] = 0.0f;

  for (int k0 = 0; k0 < K; k0 += 32) {
    const bool kFull = (k0 + 32 <= K);
    bool asyncB = false;
    if (!GENERIC && nFull && kFull) {
      // B: raw bf16 copy global->LDS via async-LDS (ASYNCcnt path).
      // 64 rows x 64B; each thread moves 2x16B. Row stride in LDS is 80B.
      int row = tid >> 1;
      int cq  = (tid & 1) * 2;
#pragma unroll
      for (int c = 0; c < 2; ++c) {
        const __bf16* src = g.Wt + (long long)(n0 + row) * K + k0 + (cq + c) * 8;
        unsigned dst = (unsigned)(unsigned long long)&Bf[row][(cq + c) * 8];
        asm volatile("global_load_async_to_lds_b128 %0, %1, off"
                     :: "v"(dst), "v"((unsigned long long)src)
                     : "memory");
      }
      asyncB = true;
    }
    // ---- stage A ----
    if (!GENERIC) {
      const int t    = k0 >> g.cinShift;
      const int cin0 = k0 - (t << g.cinShift);
      const int t3   = t / 3;
      const int dy   = t3 - 1, dx = t - t3 * 3 - 1;
      const int yy   = y0 + dy;
      const bool yok = (yy >= 0) && (yy < g.H);
      const bool xin = (dx == 0) || (dx < 0 ? (x0 >= 1) : (x0 + 64 < g.Wd));
      if (yok && xin) {
        const float* base = X + (long long)cin0 * M + (long long)yy * g.Wd + x0 + dx;
#pragma unroll
        for (int q = 0; q < 4; ++q) {
          int chunk = tid + q * 128;
          int c = chunk >> 4, r = (chunk & 15) * 4;
          f4u v = *(const f4u*)(base + (long long)c * M + r);
          store_a_frag(Af, r + 0, c, v.x);
          store_a_frag(Af, r + 1, c, v.y);
          store_a_frag(Af, r + 2, c, v.z);
          store_a_frag(Af, r + 3, c, v.w);
        }
      } else if (yok) {
#pragma unroll 1
        for (int q = 0; q < 16; ++q) {
          int idx = tid + q * 128;
          int r = idx & 63, c = idx >> 6;
          int xx = x0 + r + dx;
          float v = (xx >= 0 && xx < g.Wd)
            ? X[(long long)(cin0 + c) * M + (long long)yy * g.Wd + xx] : 0.0f;
          store_a_frag(Af, r, c, v);
        }
      } else {
#pragma unroll 1
        for (int q = 0; q < 16; ++q) {
          int idx = tid + q * 128;
          store_a_frag(Af, idx & 63, idx >> 6, 0.0f);
        }
      }
    } else {
#pragma unroll 1
      for (int q = 0; q < 16; ++q) {
        int idx = tid + q * 128;
        int r = idx & 63, c = idx >> 6;
        int k = k0 + c;
        float v = 0.0f;
        if (k < K) {
          int t = k >> g.cinShift;
          int cin = k - (t << g.cinShift);
          int t3 = t / 3;
          int yy = y0 + t3 - 1, xx = x0 + r + (t - t3 * 3 - 1);
          if (yy >= 0 && yy < g.H && xx >= 0 && xx < g.Wd)
            v = X[(long long)cin * M + (long long)yy * g.Wd + xx];
        }
        store_a_frag(Af, r, c, v);
      }
    }
    // ---- stage B (scalar bf16) when async not used ----
    if (!asyncB) {
#pragma unroll 1
      for (int q = 0; q < 16; ++q) {
        int idx = tid + q * 128;
        int k = idx & 31, n = idx >> 5;
        __bf16 v = bf_zero();
        if (k0 + k < K && n0 + n < g.Cout)
          v = g.Wt[(long long)(n0 + n) * K + k0 + k];
        Bf[n][k] = v;
      }
    } else {
      asm volatile("s_wait_asynccnt 0" ::: "memory");
    }
    __syncthreads();
    v16bf a0 = load_frag16(&Af[wy * 32      + mr][hi * 16]);
    v16bf a1 = load_frag16(&Af[wy * 32 + 16 + mr][hi * 16]);
    v16bf b0 = load_frag16(&Bf[wx * 32      + mr][hi * 16]);
    v16bf b1 = load_frag16(&Bf[wx * 32 + 16 + mr][hi * 16]);
    acc[0][0] = __builtin_amdgcn_wmma_f32_16x16x32_bf16(false, a0, false, b0, (short)0, acc[0][0], false, false);
    acc[0][1] = __builtin_amdgcn_wmma_f32_16x16x32_bf16(false, a0, false, b1, (short)0, acc[0][1], false, false);
    acc[1][0] = __builtin_amdgcn_wmma_f32_16x16x32_bf16(false, a1, false, b0, (short)0, acc[1][0], false, false);
    acc[1][1] = __builtin_amdgcn_wmma_f32_16x16x32_bf16(false, a1, false, b1, (short)0, acc[1][1], false, false);
    __syncthreads();
  }
  // vectorized epilogue: two float4 stores per lane per tile
#pragma unroll
  for (int ty = 0; ty < 2; ++ty)
#pragma unroll
    for (int tx = 0; tx < 2; ++tx) {
      int n = n0 + wx * 32 + tx * 16 + mr;
      if (n < g.Cout) {
        int mb = m0 + wy * 32 + ty * 16 + hi * 8;
        float b = g.bias ? g.bias[n] : 0.0f;
        float vv[8];
#pragma unroll
        for (int r = 0; r < 8; ++r) {
          float v = acc[ty][tx][r] + b;
          vv[r] = g.relu ? fmaxf(v, 0.0f) : v;
        }
        float4 s0 = {vv[0], vv[1], vv[2], vv[3]};
        float4 s1 = {vv[4], vv[5], vv[6], vv[7]};
        *(float4*)(Y + (long long)n * M + mb)     = s0;
        *(float4*)(Y + (long long)n * M + mb + 4) = s1;
      }
    }
}

// ---------------------------------------------------------------------------
// Elementwise / normalization / permutation kernels (fp32 VALU path)
// ---------------------------------------------------------------------------
__global__ void add2_kernel(const float* a, const float* b, float* o, long long n) {
  long long i = (long long)blockIdx.x * 256 + threadIdx.x;
  if (i < n) o[i] = a[i] + b[i];
}

__global__ void ln_channels_kernel(const float* x, const float* g, const float* bt,
                                   float* y, long long HW, long long total) {
  long long i = (long long)blockIdx.x * 256 + threadIdx.x;
  if (i >= total) return;
  long long b = i / HW, p = i - b * HW;
  const float* xb = x + b * 64 * HW + p;
  float s = 0.0f, ss = 0.0f;
#pragma unroll 4
  for (int c = 0; c < 64; ++c) { float v = xb[(long long)c * HW]; s += v; ss += v * v; }
  float mean = s * (1.0f / 64.0f);
  float var  = ss * (1.0f / 64.0f) - mean * mean;
  float rstd = rsqrtf(var + 1e-5f);
  float* yb = y + b * 64 * HW + p;
#pragma unroll 4
  for (int c = 0; c < 64; ++c)
    yb[(long long)c * HW] = (xb[(long long)c * HW] - mean) * rstd * g[c] + bt[c];
}

__global__ __launch_bounds__(128)
void ln_rows64_kernel(const float* x, const float* g, const float* bt,
                      float* y, long long rows) {
  long long row = (long long)blockIdx.x * 4 + (threadIdx.x >> 5);
  int lane = threadIdx.x & 31;
  if (row >= rows) return;
  const float* xr = x + row * 64;
  float v0 = xr[lane], v1 = xr[lane + 32];
  float s = v0 + v1, ss = v0 * v0 + v1 * v1;
#pragma unroll
  for (int m = 16; m > 0; m >>= 1) { s += __shfl_xor(s, m); ss += __shfl_xor(ss, m); }
  float mean = s * (1.0f / 64.0f);
  float var  = ss * (1.0f / 64.0f) - mean * mean;
  float rstd = rsqrtf(var + 1e-5f);
  float* yr = y + row * 64;
  yr[lane]      = (v0 - mean) * rstd * g[lane] + bt[lane];
  yr[lane + 32] = (v1 - mean) * rstd * g[lane + 32] + bt[lane + 32];
}

__global__ void dct_init_kernel(float* D) {
  int i = blockIdx.x, j = threadIdx.x;
  float v = sqrtf(2.0f / 256.0f) *
            cosf(3.14159265358979323846f * (j + 0.5f) * i / 256.0f);
  if (i == 0) v *= 0.70710678118654752f;
  D[i * 256 + j] = v;
}

__global__ void nhwc2nchw_kernel(const float* x, float* y) {
  long long i = (long long)blockIdx.x * 256 + threadIdx.x;
  if (i >= 2LL * 4 * 65536) return;
  int w = i & 255, h = (i >> 8) & 255, c = (i >> 16) & 3, b = (int)(i >> 18);
  y[i] = x[(((long long)b * 256 + h) * 256 + w) * 4 + c];
}

__global__ void cv_pre_y_kernel(const float* src, float* dst) {
  long long i = (long long)blockIdx.x * 256 + threadIdx.x;
  if (i >= 8388608LL) return;
  int w2 = i & 511; long long t = i >> 9;
  int h2 = (int)(t & 127); t >>= 7;
  int c2 = (int)(t & 63);  int b = (int)(t >> 6);
  int hh = c2 * 4 + ((h2 & 1) << 1) + (w2 & 1);
  dst[i] = src[(((long long)b * 64 + (h2 >> 1)) * 256 + hh) * 256 + (w2 >> 1)];
}
__global__ void cv_post_y_kernel(const float* src, float* dst) {
  long long i = (long long)blockIdx.x * 256 + threadIdx.x;
  if (i >= 8388608LL) return;
  int ww = i & 255, hh = (i >> 8) & 255, cc = (i >> 16) & 63, b = (int)(i >> 22);
  dst[i] = src[(((long long)b * 64 + (hh >> 2)) * 128 +
                (cc * 2 + ((hh >> 1) & 1))) * 512 + (ww * 2 + (hh & 1))];
}
__global__ void cv_pre_z_kernel(const float* src, float* dst) {
  long long i = (long long)blockIdx.x * 256 + threadIdx.x;
  if (i >= 8388608LL) return;
  int w2 = i & 511; long long t = i >> 9;
  int h2 = (int)(t & 127); t >>= 7;
  int c2 = (int)(t & 63);  int b = (int)(t >> 6);
  int ww = c2 * 4 + ((h2 & 1) << 1) + (w2 & 1);
  dst[i] = src[(((long long)b * 64 + (h2 >> 1)) * 256 + (w2 >> 1)) * 256 + ww];
}
__global__ void cv_post_z_add_kernel(const float* src, float* dst) {
  long long i = (long long)blockIdx.x * 256 + threadIdx.x;
  if (i >= 8388608LL) return;
  int ww = i & 255, hh = (i >> 8) & 255, cc = (i >> 16) & 63, b = (int)(i >> 22);
  dst[i] += src[(((long long)b * 64 + (ww >> 2)) * 128 +
                 (cc * 2 + ((ww >> 1) & 1))) * 512 + (hh * 2 + (ww & 1))];
}

__global__ void window_gather_kernel(const float* xd, float* flat) {
  long long i = (long long)blockIdx.x * 256 + threadIdx.x;
  if (i >= 8388608LL) return;
  int c = i & 63, p = (int)((i >> 6) & 255); int bw = (int)(i >> 14);
  int iw = bw & 15, ih = (bw >> 4) & 15, b = bw >> 8;
  int h = ih * 16 + (p >> 4), w = iw * 16 + (p & 15);
  flat[i] = xd[(((long long)b * 64 + c) * 256 + h) * 256 + w];
}
__global__ void window_scatter_add_kernel(const float* flat, const float* a2, float* xr) {
  long long i = (long long)blockIdx.x * 256 + threadIdx.x;
  if (i >= 8388608LL) return;
  int w = i & 255, h = (i >> 8) & 255, c = (i >> 16) & 63, b = (int)(i >> 22);
  int bw = (b * 16 + (h >> 4)) * 16 + (w >> 4);
  int p = (h & 15) * 16 + (w & 15);
  long long fi = ((long long)bw * 256 + p) * 64 + c;
  xr[i] = flat[fi] + a2[fi];
}

__global__ void mamba_conv_silu_kernel(const float* xz, const float* cw,
                                       const float* cb, float* out, long long total) {
  long long i = (long long)blockIdx.x * 256 + threadIdx.x;
  if (i >= total) return;
  int d = i & 511;
  int l = (int)((i >> 9) & 255);
  long long s = i >> 17;
  const float* base = xz + s * 256 * 1024 + d;
  float acc = cb[d];
#pragma unroll
  for (int k = 0; k < 4; ++k) {
    int ls = l + k - 3;
    if (ls >= 0) acc += base[(long long)ls * 1024] * cw[d * 4 + k];
  }
  out[i] = acc / (1.0f + __expf(-acc));
}

__global__ __launch_bounds__(512)
void mamba_scan_kernel(const float* delta, const float* dbc, const float* xsact,
                       const float* xz, const float* A_log, const float* Dp,
                       float* yout) {
  __shared__ float sBC[32];
  int s = blockIdx.x, d = threadIdx.x;
  float Ar[16];
#pragma unroll
  for (int j = 0; j < 16; ++j) Ar[j] = -__expf(A_log[d * 16 + j]);
  float h[16];
#pragma unroll
  for (int j = 0; j < 16; ++j) h[j] = 0.0f;
  float Dd = Dp[d];
  long long b48 = (long long)s * 256 * 48;
  long long b512 = (long long)s * 256 * 512;
  long long b1024 = (long long)s * 256 * 1024;
  for (int l = 0; l < 256; ++l) {
    if (d < 32) sBC[d] = dbc[b48 + (long long)l * 48 + 16 + d];
    __syncthreads();
    float dt = delta[b512 + (long long)l * 512 + d];
    float u  = xsact[b512 + (long long)l * 512 + d];
    float zz = xz[b1024 + (long long)l * 1024 + 512 + d];
    float y = 0.0f;
#pragma unroll
    for (int j = 0; j < 16; ++j) {
      h[j] = __expf(dt * Ar[j]) * h[j] + dt * sBC[j] * u;
      y += h[j] * sBC[16 + j];
    }
    float yv = y + u * Dd;
    yout[b512 + (long long)l * 512 + d] = yv * (zz / (1.0f + __expf(-zz)));
    __syncthreads();
  }
}

__global__ void final_out_kernel(const float* x_nhwc, const float* tail, float* out) {
  long long i = (long long)blockIdx.x * 256 + threadIdx.x;
  if (i >= 917504LL) return;
  int c = (int)(i % 7);
  long long t = i / 7;
  int w = (int)(t & 255), h = (int)((t >> 8) & 255), b = (int)(t >> 16);
  out[i] = (c & 1)
    ? tail[(((long long)b * 7 + c) * 256 + h) * 256 + w]
    : x_nhwc[(((long long)b * 256 + h) * 256 + w) * 4 + (c >> 1)];
}

// ---------------------------------------------------------------------------
// Host-side orchestration
// ---------------------------------------------------------------------------
static const long long HW  = 65536;
static const long long CHW = 64 * 65536;

static void gemm(hipStream_t st, const float* A, const float* B, float* C,
                 const float* bias, const float* resid,
                 int M, int N, int K, int lda, int ldb, int ldc,
                 long long sA, long long sB, long long sC, long long sR,
                 int at, int bt, int ct, int biasPerM, int act, int accum, int zc) {
  GemmArgs g{A, B, C, bias, resid, M, N, K, lda, ldb, ldc,
             sA, sB, sC, sR, ct, biasPerM, act, accum};
  dim3 grid((M + 63) / 64, (N + 63) / 64, zc);
  if (at) {
    if (bt) gemm_wmma_kernel<1, 1><<<grid, dim3(128), 0, st>>>(g);
    else    gemm_wmma_kernel<1, 0><<<grid, dim3(128), 0, st>>>(g);
  } else {
    if (bt) gemm_wmma_kernel<0, 1><<<grid, dim3(128), 0, st>>>(g);
    else    gemm_wmma_kernel<0, 0><<<grid, dim3(128), 0, st>>>(g);
  }
}

static inline dim3 elw(long long n) { return dim3((unsigned)((n + 255) / 256)); }

static void conv3x3(hipStream_t st, const float* X, const float* W, const float* bias,
                    float* Y, __bf16* wtbuf, int Cin, int Cout, int H, int Wd,
                    int Bn, int relu) {
  int K = Cin * 9;
  int cinShift = 31 - __builtin_clz((unsigned)Cin);
  conv_wt_kernel<<<elw((long long)Cout * K), 256, 0, st>>>(W, wtbuf, Cin, cinShift,
                                                           Cout * K);
  ConvArgs g{X, wtbuf, bias, Y, Cin, Cout, H, Wd, cinShift,
             (long long)Cin * H * Wd, (long long)Cout * H * Wd, relu};
  int M = H * Wd;
  dim3 grid((M + 63) / 64, (Cout + 63) / 64, Bn);
  if ((Cin & 31) == 0)
    conv3x3_wmma_kernel<0><<<grid, dim3(128), 0, st>>>(g);
  else
    conv3x3_wmma_kernel<1><<<grid, dim3(128), 0, st>>>(g);
}

static void run_crossview(hipStream_t st, const float* const* P, int pi,
                          const float* in, float* acc3,
                          float* t0, float* t1, float* t2, __bf16* wtbuf) {
  ln_channels_kernel<<<elw(2 * HW), 256, 0, st>>>(in, P[pi], P[pi + 1], t0, HW, 2 * HW);
  gemm(st, t0, P[pi + 2], t1, P[pi + 3], nullptr, (int)HW, 64, 64,
       (int)HW, 64, (int)HW, CHW, 0, CHW, 0, 1, 1, 1, 0, 0, 0, 2);
  cv_pre_y_kernel<<<elw(8388608), 256, 0, st>>>(t1, t2);
  conv3x3(st, t2, P[pi + 4], P[pi + 5], t1, wtbuf, 64, 64, 128, 512, 2, 1);
  conv3x3(st, t1, P[pi + 6], P[pi + 7], t2, wtbuf, 64, 64, 128, 512, 2, 0);
  cv_post_y_kernel<<<elw(8388608), 256, 0, st>>>(t2, acc3);
  gemm(st, t0, P[pi + 8], t1, P[pi + 9], nullptr, (int)HW, 64, 64,
       (int)HW, 64, (int)HW, CHW, 0, CHW, 0, 1, 1, 1, 0, 0, 0, 2);
  cv_pre_z_kernel<<<elw(8388608), 256, 0, st>>>(t1, t2);
  conv3x3(st, t2, P[pi + 10], P[pi + 11], t1, wtbuf, 64, 64, 128, 512, 2, 1);
  conv3x3(st, t1, P[pi + 12], P[pi + 13], t2, wtbuf, 64, 64, 128, 512, 2, 0);
  cv_post_z_add_kernel<<<elw(8388608), 256, 0, st>>>(t2, acc3);
}

extern "C" void kernel_launch(void* const* d_in, const int* in_sizes, int n_in,
                              void* d_out, int out_size, void* d_ws, size_t ws_size,
                              hipStream_t stream) {
  (void)in_sizes; (void)n_in; (void)out_size; (void)ws_size;
  const float* P[71];
  for (int i = 0; i < 71; ++i) P[i] = (const float*)d_in[i];
  const float* x_nhwc = P[0];
  // head: 2..5 ; cv1: 6..19 ; cv2: 20..33 ; intra: 34..55 ; mamba: 56..64 ;
  // fuse: 65,66 ; tail: 67..70

  float* wf = (float*)d_ws;
  long long o = 0;
  auto take = [&](long long nf) { float* p = wf + o; o += nf; return p; };
  float* dctM   = take(65536);
  __bf16* wtbuf = (__bf16*)take(64 * 576 / 2 + 64);  // bf16 transposed weights
  float* x_head = take(2 * CHW);
  float* res1   = take(2 * CHW);
  float* out1   = take(2 * CHW);
  float* outsum = take(2 * CHW);
  float* res2b  = take(2 * CHW);
  float* resb   = take(2 * CHW);
  float* t0 = take(2 * CHW);
  float* t1 = take(2 * CHW);
  float* t2 = take(2 * CHW);
  float* t3 = take(2 * CHW);
  float* t4 = take(4 * CHW);

  dct_init_kernel<<<dim3(256), 256, 0, stream>>>(dctM);

  // ---- head ----
  nhwc2nchw_kernel<<<elw(2 * 4 * HW), 256, 0, stream>>>(x_nhwc, t0);
  conv3x3(stream, t0, P[2], P[3], t1, wtbuf, 4, 64, 256, 256, 2, 1);
  conv3x3(stream, t1, P[4], P[5], x_head, wtbuf, 64, 64, 256, 256, 2, 0);

  // ---- res1 = crossview(x_head, cv1) + x_head ----
  run_crossview(stream, P, 6, x_head, t3, t0, t1, t2, wtbuf);
  add2_kernel<<<elw(2 * CHW), 256, 0, stream>>>(t3, x_head, res1, 2 * CHW);

  // ---- mamba: 4 L2-resident chunks of 32 sequences ----
  for (int ch = 0; ch < 4; ++ch) {
    const float* xs_in = res1 + (long long)ch * 32 * HW;
    float* o1 = out1 + (long long)ch * 32 * HW;
    gemm(stream, xs_in, P[56], t0, nullptr, nullptr, 256, 1024, 256,
         256, 256, 1024, 65536, 0, 262144, 0, 0, 1, 0, 0, 0, 0, 32);
    mamba_conv_silu_kernel<<<elw(32LL * 256 * 512), 256, 0, stream>>>(
        t0, P[57], P[58], t1, 32LL * 256 * 512);
    gemm(stream, t1, P[59], t4, nullptr, nullptr, 256, 48, 512,
         512, 512, 48, 131072, 0, 12288, 0, 0, 1, 0, 0, 0, 0, 32);
    gemm(stream, t4, P[60], t2, P[61], nullptr, 256, 512, 16,
         48, 16, 512, 12288, 0, 131072, 0, 0, 1, 0, 0, 3, 0, 32);
    mamba_scan_kernel<<<dim3(32), 512, 0, stream>>>(t2, t4, t1, t0,
                                                    P[62], P[63], t3);
    gemm(stream, t3, P[64], o1, nullptr, nullptr, 256, 256, 512,
         512, 512, 256, 131072, 0, 65536, 0, 0, 1, 0, 0, 0, 0, 32);
  }

  // ---- intra_slice(res1) ----
  gemm(stream, dctM, res1, t0, nullptr, nullptr, 256, 256, 256,
       256, 256, 256, 0, 65536, 65536, 0, 0, 0, 0, 0, 0, 0, 128);
  gemm(stream, t0, dctM, t1, nullptr, nullptr, 256, 256, 256,
       256, 256, 256, 65536, 0, 65536, 0, 0, 1, 0, 0, 0, 0, 128);
  ln_channels_kernel<<<elw(2 * HW), 256, 0, stream>>>(t1, P[34], P[35], t2, HW, 2 * HW);
  gemm(stream, t2, P[36], t0, P[37], nullptr, (int)HW, 64, 64,
       (int)HW, 64, (int)HW, CHW, 0, CHW, 0, 1, 1, 1, 0, 0, 0, 2);
  conv3x3(stream, t0, P[38], P[39], t2, wtbuf, 64, 64, 256, 256, 2, 0);
  conv3x3(stream, t2, P[40], P[41], t0, wtbuf, 64, 64, 256, 256, 2, 0);
  window_gather_kernel<<<elw(8388608), 256, 0, stream>>>(t0, t1);
  ln_rows64_kernel<<<elw(512LL * 256 * 32), 128, 0, stream>>>(t1, P[42], P[43],
                                                              t2, 512 * 256);
  gemm(stream, P[44], t2, t3, P[45], nullptr, 256, 64, 256,
       256, 64, 64, 0, 16384, 16384, 0, 0, 0, 0, 1, 2, 0, 512);
  gemm(stream, P[46], t3, t2, P[47], t1, 256, 64, 256,
       256, 64, 64, 0, 16384, 16384, 16384, 0, 0, 0, 1, 0, 0, 512);
  ln_rows64_kernel<<<elw(512LL * 256 * 32), 128, 0, stream>>>(t2, P[48], P[49],
                                                              t3, 512 * 256);
  gemm(stream, t3, P[50], t4, P[51], nullptr, 256, 128, 64,
       64, 64, 128, 16384, 0, 32768, 0, 0, 1, 0, 0, 2, 0, 512);
  gemm(stream, t4, P[52], t3, P[53], t2, 256, 64, 128,
       128, 128, 64, 32768, 0, 16384, 16384, 0, 1, 0, 0, 0, 0, 512);
  window_scatter_add_kernel<<<elw(8388608), 256, 0, stream>>>(t1, t3, t0);
  gemm(stream, t0, dctM, t2, nullptr, nullptr, 256, 256, 256,
       256, 256, 256, 65536, 0, 65536, 0, 0, 0, 0, 0, 0, 0, 128);
  gemm(stream, dctM, t2, t1, nullptr, nullptr, 256, 256, 256,
       256, 256, 256, 0, 65536, 65536, 0, 1, 0, 0, 0, 0, 0, 128);
  add2_kernel<<<elw(2 * CHW), 256, 0, stream>>>(out1, res1, outsum, 2 * CHW);
  gemm(stream, t1, P[54], outsum, P[55], nullptr, (int)HW, 64, 64,
       (int)HW, 64, (int)HW, CHW, 0, CHW, 0, 1, 1, 1, 0, 0, 1, 2);

  // ---- res2 = crossview(output, cv2) + output ----
  run_crossview(stream, P, 20, outsum, t3, t0, t1, t2, wtbuf);
  add2_kernel<<<elw(2 * CHW), 256, 0, stream>>>(t3, outsum, res2b, 2 * CHW);

  // ---- res = fuse_1x1(concat[res1,res2]) + x_head (split-K) ----
  gemm(stream, res1, P[65], resb, nullptr, nullptr, (int)HW, 64, 64,
       (int)HW, 128, (int)HW, CHW, 0, CHW, 0, 1, 1, 1, 0, 0, 0, 2);
  gemm(stream, res2b, P[65] + 64, resb, P[66], x_head, (int)HW, 64, 64,
       (int)HW, 128, (int)HW, CHW, 0, CHW, CHW, 1, 1, 1, 0, 0, 1, 2);

  // ---- tail ----
  conv3x3(stream, resb, P[67], P[68], t0, wtbuf, 64, 64, 256, 256, 2, 1);
  conv3x3(stream, t0, P[69], P[70], t1, wtbuf, 64, 7, 256, 256, 2, 0);

  final_out_kernel<<<elw(917504), 256, 0, stream>>>(x_nhwc, t1, (float*)d_out);
}
